// LowRankBilinearAttention_13065290515116
// MI455X (gfx1250) — matrix operand order repaired and checked by reading, not compiled, not run on hardware
//
#include <hip/hip_runtime.h>

// ---------------------------------------------------------------------------
// Problem constants
// ---------------------------------------------------------------------------
#define Bn    8
#define Pn    196
#define Ln    80
#define DIM1  2048
#define DIM2  300
#define ATT   1024

#define K2PAD 320   // DIM2 padded to multiple of 32
#define PPAD  224   // P (=K of final GEMM) padded to multiple of 32

typedef __attribute__((ext_vector_type(16))) __bf16 v16bf;
typedef __attribute__((ext_vector_type(8)))  float  v8f;

union Frag {
    v16bf v;
    unsigned int u[8];
};

__device__ __forceinline__ unsigned short f2bf(float f) {
    unsigned int x = __float_as_uint(f);
    // round-to-nearest-even f32 -> bf16
    unsigned int r = (x + 0x7FFFu + ((x >> 16) & 1u)) >> 16;
    return (unsigned short)r;
}

__device__ __forceinline__ float fast_tanh(float x) {
    float e = __expf(2.0f * x);          // v_exp_f32 (TRANS)
    return 1.0f - 2.0f / (e + 1.0f);
}

// ---------------------------------------------------------------------------
// Kernel 0: w = Wh @ Wt  (ATT floats),  c = bh.Wt + bt   (one wave per row)
// ---------------------------------------------------------------------------
__global__ void k_wfold(const float* __restrict__ Wh, const float* __restrict__ Wt,
                        const float* __restrict__ bh, const float* __restrict__ bt,
                        float* __restrict__ wvec, float* __restrict__ cval) {
    int gw   = (blockIdx.x * blockDim.x + threadIdx.x) >> 5;
    int lane = threadIdx.x & 31;
    if (gw < ATT) {
        float acc = 0.f;
        for (int j = lane; j < ATT; j += 32) acc += Wh[gw * ATT + j] * Wt[j];
        for (int off = 16; off; off >>= 1) acc += __shfl_xor(acc, off, 32);
        if (lane == 0) wvec[gw] = acc;
    }
    if (blockIdx.x == 0 && threadIdx.x < 32) {
        float acc = 0.f;
        for (int j = lane; j < ATT; j += 32) acc += bh[j] * Wt[j];
        for (int off = 16; off; off >>= 1) acc += __shfl_xor(acc, off, 32);
        if (lane == 0) cval[0] = acc + bt[0];
    }
}

// ---------------------------------------------------------------------------
// f32 (R x Ks) row-major  ->  bf16 (R x Kp) row-major, zero-padded K
// ---------------------------------------------------------------------------
__global__ void k_cvt_rows(const float* __restrict__ src, unsigned short* __restrict__ dst,
                           int R, int Ks, int Kp) {
    long long i = (long long)blockIdx.x * blockDim.x + threadIdx.x;
    long long total = (long long)R * Kp;
    if (i >= total) return;
    int k = (int)(i % Kp);
    long long r = i / Kp;
    float v = (k < Ks) ? src[r * Ks + k] : 0.f;
    dst[i] = f2bf(v);
}

// ---------------------------------------------------------------------------
// batched: src[b] = (K x N) f32 row-major  ->  dst[b] = (N x Kp) bf16 (transposed)
// ---------------------------------------------------------------------------
__global__ void k_cvt_transpose(const float* __restrict__ src, unsigned short* __restrict__ dst,
                                int K, int N, int Kp, long long sSrc, long long sDst) {
    long long i = (long long)blockIdx.x * blockDim.x + threadIdx.x;
    long long per = (long long)N * Kp;
    if (i >= per) return;
    int b = blockIdx.z;
    int k = (int)(i % Kp);
    long long n = i / Kp;
    float v = (k < K) ? src[(long long)b * sSrc + (long long)k * N + n] : 0.f;
    dst[(long long)b * sDst + i] = f2bf(v);
}

// ---------------------------------------------------------------------------
// Batched GEMM: C[b] (MxN f32) = A[b] (MxKp bf16, row-major) * Bt[b]^T
//   Bt[b] is (N x Kp) bf16 row-major (i.e., B pre-transposed, K contiguous).
// One wave -> 16x64 of C (one A fragment reused over 4 WMMA N-tiles).
// 256 threads = 8 waves per block.
// ---------------------------------------------------------------------------
__global__ void k_gemm_bf16(const unsigned short* __restrict__ A,
                            const unsigned short* __restrict__ Bt,
                            float* __restrict__ C,
                            int M, int N, int Kp,
                            long long sA, long long sB, long long sC) {
    const int wave = threadIdx.x >> 5;
    const int lane = threadIdx.x & 31;
    const int bz = blockIdx.z;
    A  += (long long)bz * sA;
    Bt += (long long)bz * sB;
    C  += (long long)bz * sC;

    const int tilesM  = M >> 4;
    const int groupsN = N >> 6;
    const int gw = blockIdx.x * 8 + wave;
    if (gw >= tilesM * groupsN) return;
    const int mt = gw % tilesM;
    const int ng = gw / tilesM;

    v8f acc0 = {}, acc1 = {}, acc2 = {}, acc3 = {};

    const int m    = mt * 16 + (lane & 15);
    const int kh8  = (lane >> 4) * 8;     // unused: A also uses kh16
    const int kh16 = (lane >> 4) * 16;    // B: lane-half K offset
    const int nlo  = (lane & 15);
    const unsigned short* arow = A + (long long)m * Kp;

    for (int kk = 0; kk < Kp; kk += 32) {
        // A fragment: 16x32 bf16, 16 consecutive K per lane half
        Frag fa;
#pragma unroll
        for (int j = 0; j < 8; ++j) {
            int k = kh16 + (j << 1);
            fa.u[j] = *(const unsigned int*)(arow + kk + k);
        }
#pragma unroll
        for (int t = 0; t < 4; ++t) {
            // B fragment: lane holds column n, 16 consecutive K values
            const unsigned short* brow =
                Bt + (long long)(ng * 64 + t * 16 + nlo) * Kp + kk + kh16;
            const uint4* bp = (const uint4*)brow;
            uint4 lo = bp[0], hi = bp[1];
            Frag fb;
            fb.u[0] = lo.x; fb.u[1] = lo.y; fb.u[2] = lo.z; fb.u[3] = lo.w;
            fb.u[4] = hi.x; fb.u[5] = hi.y; fb.u[6] = hi.z; fb.u[7] = hi.w;
            v8f& acc = (t == 0) ? acc0 : (t == 1) ? acc1 : (t == 2) ? acc2 : acc3;
            acc = __builtin_amdgcn_wmma_f32_16x16x32_bf16(
                false, fa.v, false, fb.v, (short)0, acc, false, false);
        }
    }

#pragma unroll
    for (int t = 0; t < 4; ++t) {
        v8f acc = (t == 0) ? acc0 : (t == 1) ? acc1 : (t == 2) ? acc2 : acc3;
        int n = ng * 64 + t * 16 + nlo;
#pragma unroll
        for (int r = 0; r < 8; ++r) {
            int mr = mt * 16 + (lane >> 4) * 8 + r;   // C/D layout: VGPR r -> M=r / M=8+r
            C[(long long)mr * N + n] = acc[r];
        }
    }
}

// ---------------------------------------------------------------------------
// Fused score + softmax. One 256-thread block per (b,l).
//   score[p] = sum_a tanh(a1[b,p,a]*a2[b,l,a]) * w[a] + c
//   alpha = softmax_p(score); writes f32 alpha to out and bf16 padded alpha to ws
// Fully 128-bit vectorized inner loop: each lane owns 4 consecutive elements.
// ---------------------------------------------------------------------------
__global__ void k_score(const float* __restrict__ a1, const float* __restrict__ a2,
                        const float* __restrict__ wvec, const float* __restrict__ cval,
                        float* __restrict__ alpha_out, unsigned short* __restrict__ alphapad) {
    __shared__ __align__(16) float s_a2[ATT];
    __shared__ __align__(16) float s_w[ATT];
    __shared__ float s_score[Pn];
    __shared__ float s_red[256];

    const int bl = blockIdx.x;          // b*L + l
    const int b  = bl / Ln;
    const int tid = threadIdx.x;
    const int wave = tid >> 5, lane = tid & 31;

    // stage a2 row and w with b128 copies (256 threads x 1 float4 each)
    {
        const float4* a2v = (const float4*)(a2 + (long long)bl * ATT);
        const float4* wv  = (const float4*)wvec;
        ((float4*)s_a2)[tid] = a2v[tid];
        ((float4*)s_w)[tid]  = wv[tid];
    }
    __syncthreads();

    const float c = cval[0];
    for (int p = wave; p < Pn; p += 8) {
        const float4* row = (const float4*)(a1 + ((long long)(b * Pn + p)) * ATT);
        float acc = 0.f;
#pragma unroll
        for (int it = 0; it < ATT / 128; ++it) {       // 8 iters, 4 elts/lane/iter
            const int i4 = it * 32 + lane;             // float4 index
            float4 va = row[i4];                       // global_load_b128
            float4 aa = ((const float4*)s_a2)[i4];     // ds_load_b128
            float4 ww = ((const float4*)s_w)[i4];      // ds_load_b128
            acc += fast_tanh(va.x * aa.x) * ww.x;
            acc += fast_tanh(va.y * aa.y) * ww.y;
            acc += fast_tanh(va.z * aa.z) * ww.z;
            acc += fast_tanh(va.w * aa.w) * ww.w;
        }
        for (int off = 16; off; off >>= 1) acc += __shfl_xor(acc, off, 32);
        if (lane == 0) s_score[p] = acc + c;
    }
    __syncthreads();

    // softmax over P = 196
    float v = (tid < Pn) ? s_score[tid] : -3.0e38f;
    s_red[tid] = v;
    __syncthreads();
    for (int s = 128; s > 0; s >>= 1) {
        if (tid < s) s_red[tid] = fmaxf(s_red[tid], s_red[tid + s]);
        __syncthreads();
    }
    const float mx = s_red[0];
    __syncthreads();
    float e = (tid < Pn) ? __expf(v - mx) : 0.f;
    s_red[tid] = e;
    __syncthreads();
    for (int s = 128; s > 0; s >>= 1) {
        if (tid < s) s_red[tid] += s_red[tid + s];
        __syncthreads();
    }
    const float inv = 1.0f / s_red[0];
    if (tid < Pn) {
        float al = e * inv;
        alpha_out[(long long)bl * Pn + tid] = al;
        alphapad[(long long)bl * PPAD + tid] = f2bf(al);
    } else if (tid < PPAD) {
        alphapad[(long long)bl * PPAD + tid] = 0;   // zero K-padding
    }
}

// ---------------------------------------------------------------------------
// Workspace layout (bytes)
// ---------------------------------------------------------------------------
static const size_t OFF_W    = 0;                                   // 1024 f32
static const size_t OFF_C    = 4096;                                // 1 f32
static const size_t OFF_A1   = 8192;                                // Bn*Pn*ATT f32
static const size_t OFF_A2   = OFF_A1 + (size_t)Bn*Pn*ATT*4;        // Bn*Ln*ATT f32
static const size_t OFF_X1B  = OFF_A2 + (size_t)Bn*Ln*ATT*4;        // Bn*Pn*DIM1 bf16
static const size_t OFF_W1T  = OFF_X1B + (size_t)Bn*Pn*DIM1*2;      // ATT*DIM1 bf16
static const size_t OFF_X2B  = OFF_W1T + (size_t)ATT*DIM1*2;        // Bn*Ln*K2PAD bf16
static const size_t OFF_W2T  = OFF_X2B + (size_t)Bn*Ln*K2PAD*2;     // ATT*K2PAD bf16
static const size_t OFF_X1T  = OFF_W2T + (size_t)ATT*K2PAD*2;       // Bn*DIM1*PPAD bf16
static const size_t OFF_ALP  = OFF_X1T + (size_t)Bn*DIM1*PPAD*2;    // Bn*Ln*PPAD bf16

extern "C" void kernel_launch(void* const* d_in, const int* in_sizes, int n_in,
                              void* d_out, int out_size, void* d_ws, size_t ws_size,
                              hipStream_t stream) {
    const float* x1 = (const float*)d_in[0];
    const float* x2 = (const float*)d_in[1];
    const float* W1 = (const float*)d_in[2];
    const float* W2 = (const float*)d_in[3];
    const float* Wh = (const float*)d_in[4];
    const float* bh = (const float*)d_in[5];
    const float* Wt = (const float*)d_in[6];
    const float* bt = (const float*)d_in[7];

    char* ws = (char*)d_ws;
    float*          wvec  = (float*)(ws + OFF_W);
    float*          cval  = (float*)(ws + OFF_C);
    float*          a1    = (float*)(ws + OFF_A1);
    float*          a2    = (float*)(ws + OFF_A2);
    unsigned short* x1b   = (unsigned short*)(ws + OFF_X1B);
    unsigned short* w1t   = (unsigned short*)(ws + OFF_W1T);
    unsigned short* x2b   = (unsigned short*)(ws + OFF_X2B);
    unsigned short* w2t   = (unsigned short*)(ws + OFF_W2T);
    unsigned short* x1t   = (unsigned short*)(ws + OFF_X1T);
    unsigned short* alp   = (unsigned short*)(ws + OFF_ALP);

    float* label_out = (float*)d_out;                         // Bn*Ln*DIM1
    float* alpha_out = label_out + (size_t)Bn * Ln * DIM1;    // Bn*Ln*Pn

    // 0) fold Wh@Wt and scalar c
    k_wfold<<<128, 256, 0, stream>>>(Wh, Wt, bh, bt, wvec, cval);

    // 1) conversions
    {
        long long tot = (long long)Bn * Pn * DIM1;
        k_cvt_rows<<<(unsigned)((tot + 255) / 256), 256, 0, stream>>>(x1, x1b, Bn * Pn, DIM1, DIM1);
    }
    {
        long long tot = (long long)Bn * Ln * K2PAD;
        k_cvt_rows<<<(unsigned)((tot + 255) / 256), 256, 0, stream>>>(x2, x2b, Bn * Ln, DIM2, K2PAD);
    }
    {
        long long per = (long long)ATT * DIM1;    // W1t: (ATT x DIM1)
        dim3 g((unsigned)((per + 255) / 256), 1, 1);
        k_cvt_transpose<<<g, 256, 0, stream>>>(W1, w1t, DIM1, ATT, DIM1, 0, 0);
    }
    {
        long long per = (long long)ATT * K2PAD;   // W2t: (ATT x K2PAD)
        dim3 g((unsigned)((per + 255) / 256), 1, 1);
        k_cvt_transpose<<<g, 256, 0, stream>>>(W2, w2t, DIM2, ATT, K2PAD, 0, 0);
    }
    {
        long long per = (long long)DIM1 * PPAD;   // x1t[b]: (DIM1 x PPAD)
        dim3 g((unsigned)((per + 255) / 256), 1, Bn);
        k_cvt_transpose<<<g, 256, 0, stream>>>(x1, x1t, Pn, DIM1, PPAD,
                                               (long long)Pn * DIM1, (long long)DIM1 * PPAD);
    }

    // 2) a1 = x1 @ W1 : M=1568, N=1024, K=2048   (98*16 = 1568 wave-tiles)
    {
        int waves = (Bn * Pn / 16) * (ATT / 64);
        dim3 g((unsigned)((waves + 7) / 8), 1, 1);
        k_gemm_bf16<<<g, 256, 0, stream>>>(x1b, w1t, a1, Bn * Pn, ATT, DIM1, 0, 0, 0);
    }
    // 3) a2 = x2 @ W2 : M=640, N=1024, K=320
    {
        int waves = (Bn * Ln / 16) * (ATT / 64);
        dim3 g((unsigned)((waves + 7) / 8), 1, 1);
        k_gemm_bf16<<<g, 256, 0, stream>>>(x2b, w2t, a2, Bn * Ln, ATT, K2PAD, 0, 0, 0);
    }
    // 4) fused score + softmax
    k_score<<<Bn * Ln, 256, 0, stream>>>(a1, a2, wvec, cval, alpha_out, alp);

    // 5) label_repr[b] = alpha[b] @ x1[b] : M=80, N=2048, K=224, batched over Bn
    {
        int waves = (Ln / 16) * (DIM1 / 64);      // 5 * 32 = 160
        dim3 g((unsigned)((waves + 7) / 8), 1, Bn);
        k_gemm_bf16<<<g, 256, 0, stream>>>(alp, x1t, label_out, Ln, DIM1, PPAD,
                                           (long long)Ln * PPAD,
                                           (long long)DIM1 * PPAD,
                                           (long long)Ln * DIM1);
    }
}
